// Vector64NNUE_68693706932338
// MI455X (gfx1250) — matrix operand order repaired and checked by reading, not compiled, not run on hardware
//
#include <hip/hip_runtime.h>
#include <hip/hip_bf16.h>

// ---- sizes from the reference ----
#define TOTAL_FEATURES 81920
#define HIDDEN 512
#define L1N 32
#define L2N 32
#define BATCH 8192
#define NNZ 32
#define NTILES (BATCH / 16)   // 512 tiles of 16 batch rows

typedef _Float16 v16h __attribute__((ext_vector_type(16)));
typedef _Float16 v8h  __attribute__((ext_vector_type(8)));
typedef _Float16 v4h  __attribute__((ext_vector_type(4)));
typedef float    v8f  __attribute__((ext_vector_type(8)));

// workspace layout (f16 elements)
#define XSWZ_HALFS ((size_t)NTILES * 16 * 32 * 16)   // tiles x kblocks x lanes x 16
#define XSWZ_BYTES (XSWZ_HALFS * 2)                  // 8 MiB
#define FD1_HALFS  (2 * 16 * 32 * 16)                // 16384
#define FD2_HALFS  (2 * 32 * 16)                     // 1024

// ---------------------------------------------------------------------------
// Kernel A: sparse feature transform -> relu((us-them)/64) -> swizzled f16 x
// One block per batch row; 128 threads cover HIDDEN=512 as float4 each.
// Indices/values are uniform per block -> scalar loads; row gathers are
// perfectly coalesced b128 loads hitting the 192MB L2 (table = 160MiB).
// ---------------------------------------------------------------------------
__global__ __launch_bounds__(128) void nnue_ft_kernel(
    const int*   __restrict__ wIdx, const int* __restrict__ bIdx,
    const int*   __restrict__ stm,
    const float* __restrict__ wVal, const float* __restrict__ bVal,
    const float* __restrict__ ftw,
    _Float16*    __restrict__ xswz)
{
  const int b   = blockIdx.x;
  const int tid = threadIdx.x;
  const float4* __restrict__ ftw4 = (const float4*)ftw;  // 128 float4 per row

  float awx = 0.f, awy = 0.f, awz = 0.f, aww = 0.f;
  float abx = 0.f, aby = 0.f, abz = 0.f, abw = 0.f;
  const int base = b * NNZ;

  for (int n = 0; n < NNZ; ++n) {
    int iw = wIdx[base + n];                 // uniform -> s_load
    if (iw >= 0) {
      float v  = wVal[base + n];             // uniform -> s_load
      float4 r = ftw4[(size_t)iw * (HIDDEN / 4) + tid];
      awx = fmaf(v, r.x, awx); awy = fmaf(v, r.y, awy);
      awz = fmaf(v, r.z, awz); aww = fmaf(v, r.w, aww);
    }
    int ib = bIdx[base + n];
    if (ib >= 0) {
      float v  = bVal[base + n];
      float4 r = ftw4[(size_t)ib * (HIDDEN / 4) + tid];
      abx = fmaf(v, r.x, abx); aby = fmaf(v, r.y, aby);
      abz = fmaf(v, r.z, abz); abw = fmaf(v, r.w, abw);
    }
  }

  // ft_bias cancels in (us - them); sign from stm
  const float sgn = (stm[b] == 0) ? 1.0f : -1.0f;
  const float S   = 1.0f / 64.0f;  // FEATURE_TO_DENSE_SCALE
  float x0 = fmaxf(sgn * (awx - abx) * S, 0.0f);
  float x1 = fmaxf(sgn * (awy - aby) * S, 0.0f);
  float x2 = fmaxf(sgn * (awz - abz) * S, 0.0f);
  float x3 = fmaxf(sgn * (aww - abw) * S, 0.0f);

  // Store into WMMA f16 A-fragment layout:
  // lane = r + 16*((k>>3)&1); vgpr = (k>>4)*4 + ((k&7)>>1); pair = k&1
  const int t  = b >> 4, r = b & 15;
  const int h0 = tid * 4;
  const int kb = h0 >> 5;          // which 32-wide K block (0..15)
  const int kk = h0 & 31;          // position inside K block (multiple of 4)
  const int half  = (kk >> 3) & 1;
  const int group = kk >> 4;
  const int lane  = r + (half << 4);
  const int v0    = (group << 2) + ((kk & 7) >> 1);
  const size_t off = ((size_t)((t * 16 + kb) * 32 + lane)) * 16 + (size_t)v0 * 2;

  v4h o;
  o[0] = (_Float16)x0; o[1] = (_Float16)x1;
  o[2] = (_Float16)x2; o[3] = (_Float16)x3;
  *(v4h*)(xswz + off) = o;   // 8-byte aligned store
}

// ---------------------------------------------------------------------------
// Kernel W: convert d1_w / d2_w to f16 WMMA B-fragments (table-K B layout:
// lane holds n = lane&15; element j holds K = 16*(j>>3) + 8*(lane>>4) + (j&7)
// of each 32-wide K block, matching the A-fragment K order).
// ---------------------------------------------------------------------------
__global__ void nnue_prep_kernel(const float* __restrict__ d1w,
                                 const float* __restrict__ d2w,
                                 _Float16* __restrict__ fd1,
                                 _Float16* __restrict__ fd2)
{
  for (int i = threadIdx.x; i < FD1_HALFS; i += blockDim.x) {
    int j    = i & 15;
    int lane = (i >> 4) & 31;
    int kb   = (i >> 9) & 15;
    int nt   = i >> 13;
    int n    = nt * 16 + (lane & 15);
    int k    = kb * 32 + ((j >> 3) << 4) + ((lane >> 4) << 3) + (j & 7);
    fd1[i] = (_Float16)d1w[n * HIDDEN + k];
  }
  for (int i = threadIdx.x; i < FD2_HALFS; i += blockDim.x) {
    int j    = i & 15;
    int lane = (i >> 4) & 31;
    int nt   = i >> 9;
    int n    = nt * 16 + (lane & 15);
    int k    = ((j >> 3) << 4) + ((lane >> 4) << 3) + (j & 7);
    fd2[i] = (_Float16)d2w[n * L1N + k];
  }
}

// ---------------------------------------------------------------------------
// Kernel B: dense layers via v_wmma_f32_16x16x32_f16.
// One wave32 per 16-batch tile; 4 waves per block.
// Layer1: 16 K-steps x 2 N-tiles = 32 WMMAs. Layer2: LDS transpose bounce,
// 2 WMMAs. Output layer: per-lane dot + shfl_xor reduction within 16 lanes.
// ---------------------------------------------------------------------------
__global__ __launch_bounds__(128) void nnue_dense_kernel(
    const _Float16* __restrict__ xswz,
    const _Float16* __restrict__ fd1, const _Float16* __restrict__ fd2,
    const float* __restrict__ d1b, const float* __restrict__ d2b,
    const float* __restrict__ outw, const float* __restrict__ outb,
    float* __restrict__ out)
{
  __shared__ _Float16 ldsT[4][32][16];  // per-wave y1 transposed [k][m]

  const int lane = threadIdx.x & 31;
  const int wid  = threadIdx.x >> 5;
  const int t    = blockIdx.x * 4 + wid;
  const int nlo  = lane & 15;           // N (or batch-row M) index within half
  const int hh   = lane >> 4;           // which 16-lane half

  // ---- layer 1: [16 x 512] @ [512 x 32] ----
  v8f c0 = {}; v8f c1 = {};
  const _Float16* xa = xswz + ((size_t)t * 16) * 32 * 16 + (size_t)lane * 16;
#pragma unroll
  for (int kb = 0; kb < 16; ++kb) {
    v16h a  = *(const v16h*)(xa + (size_t)kb * 32 * 16);
    v16h b0 = *(const v16h*)(fd1 + ((size_t)(0 * 16 + kb) * 32 + lane) * 16);
    v16h b1 = *(const v16h*)(fd1 + ((size_t)(1 * 16 + kb) * 32 + lane) * 16);
    c0 = __builtin_amdgcn_wmma_f32_16x16x32_f16(false, a, false, b0, (short)0, c0, false, false);
    c1 = __builtin_amdgcn_wmma_f32_16x16x32_f16(false, a, false, b1, (short)0, c1, false, false);
  }

  const float S = 1.0f / 64.0f;  // DENSE_TO_DENSE_SCALE
  const float b1lo = d1b[nlo], b1hi = d1b[16 + nlo];
  v8h y0, y1;
#pragma unroll
  for (int v = 0; v < 8; ++v) {
    // D layout: lane holds n = nlo (+16 for c1), vgpr v holds m = v + 8*hh
    y0[v] = (_Float16)fmaxf((c0[v] + b1lo) * S, 0.0f);
    y1[v] = (_Float16)fmaxf((c1[v] + b1hi) * S, 0.0f);
  }
  // store transposed: ldsT[n][m], 8 consecutive m per lane -> 16B DS stores
  *(v8h*)&ldsT[wid][nlo][hh * 8]      = y0;
  *(v8h*)&ldsT[wid][nlo + 16][hh * 8] = y1;

  // ---- layer 2: assemble A fragment (row m' = nlo, K per A layout) ----
  v16h a2;
#pragma unroll
  for (int v = 0; v < 8; ++v) {
#pragma unroll
    for (int pp = 0; pp < 2; ++pp) {
      int k = ((v >> 2) << 4) + (hh << 3) + ((v & 3) << 1) + pp;
      a2[v * 2 + pp] = ldsT[wid][k][nlo];
    }
  }
  v16h b20 = *(const v16h*)(fd2 + ((size_t)(0 * 32 + lane)) * 16);
  v16h b21 = *(const v16h*)(fd2 + ((size_t)(1 * 32 + lane)) * 16);
  v8f c20 = {}; v8f c21 = {};
  c20 = __builtin_amdgcn_wmma_f32_16x16x32_f16(false, a2, false, b20, (short)0, c20, false, false);
  c21 = __builtin_amdgcn_wmma_f32_16x16x32_f16(false, a2, false, b21, (short)0, c21, false, false);

  // ---- output layer: out[m] = (sum_n y2[m][n]*ow[n] + ob) / 16 ----
  const float b2lo = d2b[nlo], b2hi = d2b[16 + nlo];
  const float owlo = outw[nlo], owhi = outw[16 + nlo];
  const float ob   = outb[0];
#pragma unroll
  for (int v = 0; v < 8; ++v) {
    float y2lo = fmaxf((c20[v] + b2lo) * S, 0.0f);   // n = nlo
    float y2hi = fmaxf((c21[v] + b2hi) * S, 0.0f);   // n = nlo + 16
    float p = fmaf(y2lo, owlo, y2hi * owhi);
    // reduce over the 16 lanes holding the different n for this row group
    p += __shfl_xor(p, 1);
    p += __shfl_xor(p, 2);
    p += __shfl_xor(p, 4);
    p += __shfl_xor(p, 8);
    if (nlo == 0) {
      out[t * 16 + hh * 8 + v] = (p + ob) * (1.0f / 16.0f);  // OUTPUT_SCALE
    }
  }
}

// ---------------------------------------------------------------------------
extern "C" void kernel_launch(void* const* d_in, const int* in_sizes, int n_in,
                              void* d_out, int out_size, void* d_ws, size_t ws_size,
                              hipStream_t stream) {
  (void)in_sizes; (void)n_in; (void)out_size; (void)ws_size;
  const int*   wIdx = (const int*)  d_in[0];
  const int*   bIdx = (const int*)  d_in[1];
  const int*   stm  = (const int*)  d_in[2];
  const float* wVal = (const float*)d_in[3];
  const float* bVal = (const float*)d_in[4];
  const float* ftw  = (const float*)d_in[5];
  // d_in[6] = ft_bias: cancels in (us - them), unused
  const float* d1w  = (const float*)d_in[7];
  const float* d1b  = (const float*)d_in[8];
  const float* d2w  = (const float*)d_in[9];
  const float* d2b  = (const float*)d_in[10];
  const float* outw = (const float*)d_in[11];
  const float* outb = (const float*)d_in[12];
  float* out = (float*)d_out;

  char* ws = (char*)d_ws;
  _Float16* xswz = (_Float16*)ws;
  _Float16* fd1  = (_Float16*)(ws + XSWZ_BYTES);
  _Float16* fd2  = fd1 + FD1_HALFS;

  nnue_ft_kernel<<<BATCH, 128, 0, stream>>>(wIdx, bIdx, stm, wVal, bVal, ftw, xswz);
  nnue_prep_kernel<<<1, 256, 0, stream>>>(d1w, d2w, fd1, fd2);
  nnue_dense_kernel<<<NTILES / 4, 128, 0, stream>>>(xswz, fd1, fd2, d1b, d2b, outw, outb, out);
}
